// Block_49254684950688
// MI455X (gfx1250) — matrix operand order repaired and checked by reading, not compiled, not run on hardware
//
#include <hip/hip_runtime.h>
#include <hip/hip_bf16.h>
#include <math.h>

typedef _Float16 f16;
typedef __attribute__((ext_vector_type(8)))  _Float16 v8h;
typedef __attribute__((ext_vector_type(16))) _Float16 v16h;
typedef __attribute__((ext_vector_type(8)))  float    v8f;

#define NTOK   197
#define NPAD   224
#define NHEAD  16
#define HDIM   64
#define DMODEL 1024
#define DFF    4096
#define BATCH  64
#define MROWS  (BATCH * NTOK)   // 12608

static __device__ __forceinline__ v16h mk16(v8h lo, v8h hi) {
  v16h r;
#pragma unroll
  for (int i = 0; i < 8; ++i) { r[i] = lo[i]; r[i + 8] = hi[i]; }
  return r;
}

static __device__ __forceinline__ v8f wmma_f16(v16h a, v16h b, v8f c) {
  // D = A(16x32,f16) x B(32x16,f16) + C(16x16,f32)
  return __builtin_amdgcn_wmma_f32_16x16x32_f16(
      /*neg_a=*/false, a, /*neg_b=*/false, b,
      /*c_mod=*/(short)0, c, /*reuse_a=*/false, /*reuse_b=*/false);
}

// ---------------------------------------------------------------- utilities

__global__ __launch_bounds__(256) void f32_to_f16_kernel(
    const float* __restrict__ src, f16* __restrict__ dst, int n) {
  for (int i = blockIdx.x * 256 + threadIdx.x; i < n; i += gridDim.x * 256)
    dst[i] = (f16)src[i];
}

__global__ __launch_bounds__(256) void qkv_bias_kernel(
    const float* __restrict__ qb, const float* __restrict__ vb,
    float* __restrict__ out) {
  int i = blockIdx.x * 256 + threadIdx.x;
  if (i < 3 * DMODEL) {
    float v = 0.f;
    if (i < DMODEL)            v = qb[i];
    else if (i >= 2 * DMODEL)  v = vb[i - 2 * DMODEL];
    out[i] = v;
  }
}

// LayerNorm over last dim (1024), fp32 in -> f16 out. One block per row.
__global__ __launch_bounds__(256) void ln_f16_kernel(
    const float* __restrict__ x, const float* __restrict__ g,
    const float* __restrict__ bta, f16* __restrict__ out) {
  __shared__ float red[16];
  const int row = blockIdx.x;
  const float* xr = x + (size_t)row * DMODEL;
  float v[4]; float s = 0.f, sq = 0.f;
#pragma unroll
  for (int i = 0; i < 4; ++i) {
    v[i] = xr[threadIdx.x + i * 256];
    s += v[i]; sq += v[i] * v[i];
  }
#pragma unroll
  for (int m = 1; m < 32; m <<= 1) {
    s  += __shfl_xor(s,  m, 32);
    sq += __shfl_xor(sq, m, 32);
  }
  const int wave = threadIdx.x >> 5;
  if ((threadIdx.x & 31) == 0) { red[wave] = s; red[8 + wave] = sq; }
  __syncthreads();
  s = 0.f; sq = 0.f;
#pragma unroll
  for (int w = 0; w < 8; ++w) { s += red[w]; sq += red[8 + w]; }
  const float mu  = s * (1.f / DMODEL);
  const float var = sq * (1.f / DMODEL) - mu * mu;
  const float rs  = rsqrtf(var + 1e-5f);
#pragma unroll
  for (int i = 0; i < 4; ++i) {
    int c = threadIdx.x + i * 256;
    out[(size_t)row * DMODEL + c] = (f16)((v[i] - mu) * rs * g[c] + bta[c]);
  }
}

// ----------------------------------------------------- generic WMMA GEMM
// C[M,N] = A[M,K] (f16, row-major) x Bw[N,K]^T (f16, row-major = W matrix)
// Block tile 128x64, 8 waves, each wave 32x32 (2x2 WMMA tiles), K step 32.
// Double-buffered LDS staging + global prefetch two tiles ahead.
__global__ __launch_bounds__(256) void gemm_wmma_kernel(
    const f16* __restrict__ A, const f16* __restrict__ Bw,
    const float* __restrict__ bias, const float* __restrict__ residual,
    float* __restrict__ outF32, f16* __restrict__ outF16,
    int M, int N, int K, int gelu) {
  __shared__ f16 As[2][128 * 32];
  __shared__ f16 Bs[2][64 * 32];
  const int tid  = threadIdx.x;
  const int lane = tid & 31, wave = tid >> 5;
  const int hl = lane >> 4, li = lane & 15;
  const int bm = blockIdx.y * 128, bn = blockIdx.x * 64;
  const int wm = (wave & 3) * 32, wn = (wave >> 2) * 32;
  v8f acc[2][2] = {};

  const int ar = tid >> 1, akc = (tid & 1) * 16;   // A: 2 thr/row, 16 halves each
  const int br = tid >> 2, bkc = (tid & 3) * 8;    // B: 4 thr/row,  8 halves each
  const int  gra  = bm + ar;
  const bool aval = gra < M;                        // constant over the K loop
  const f16* aSrc = A + (size_t)(aval ? gra : 0) * K + akc;
  const f16* bSrc = Bw + (size_t)(bn + br) * K + bkc;

  // preload tile 0 into buffer 0
  {
    v8h a0 = {}, a1 = {};
    if (aval) { a0 = *(const v8h*)aSrc; a1 = *(const v8h*)(aSrc + 8); }
    *(v8h*)(&As[0][ar * 32 + akc])     = a0;
    *(v8h*)(&As[0][ar * 32 + akc + 8]) = a1;
    *(v8h*)(&Bs[0][br * 32 + bkc])     = *(const v8h*)bSrc;
  }
  __syncthreads();

  const int nIter = K >> 5;
  for (int it = 0; it < nIter; ++it) {
    const int cur = it & 1;
    // kick off global loads for tile it+1 while computing on tile it
    v8h na0 = {}, na1 = {}, nb0 = {};
    const bool have_next = (it + 1) < nIter;
    if (have_next) {
      const int ko = (it + 1) * 32;
      if (aval) {
        na0 = *(const v8h*)(aSrc + ko);
        na1 = *(const v8h*)(aSrc + ko + 8);
      }
      nb0 = *(const v8h*)(bSrc + ko);
      if ((it + 2) < nIter) {               // pull tile it+2 toward the WGP
        __builtin_prefetch(aSrc + (it + 2) * 32, 0, 1);
        __builtin_prefetch(bSrc + (it + 2) * 32, 0, 1);
      }
    }
    // fragments from current LDS buffer
    v16h af[2], bf[2];
#pragma unroll
    for (int sm = 0; sm < 2; ++sm) {
      const f16* p = &As[cur][(wm + sm * 16 + li) * 32 + hl * 8];
      af[sm] = mk16(*(const v8h*)p, *(const v8h*)(p + 16));
    }
#pragma unroll
    for (int sn = 0; sn < 2; ++sn) {
      const f16* p = &Bs[cur][(wn + sn * 16 + li) * 32 + hl * 8];
      bf[sn] = mk16(*(const v8h*)p, *(const v8h*)(p + 16));
    }
#pragma unroll
    for (int sm = 0; sm < 2; ++sm)
#pragma unroll
      for (int sn = 0; sn < 2; ++sn)
        acc[sm][sn] = wmma_f16(af[sm], bf[sn], acc[sm][sn]);
    // stage tile it+1 into the other buffer
    if (have_next) {
      *(v8h*)(&As[cur ^ 1][ar * 32 + akc])     = na0;
      *(v8h*)(&As[cur ^ 1][ar * 32 + akc + 8]) = na1;
      *(v8h*)(&Bs[cur ^ 1][br * 32 + bkc])     = nb0;
    }
    __syncthreads();
  }

  // Epilogue: C layout per lane -> M = r + 8*(lane/16), N = lane%16
#pragma unroll
  for (int sm = 0; sm < 2; ++sm) {
#pragma unroll
    for (int sn = 0; sn < 2; ++sn) {
      int n = bn + wn + sn * 16 + li;
      float bv = bias ? bias[n] : 0.f;
#pragma unroll
      for (int r = 0; r < 8; ++r) {
        int m = bm + wm + sm * 16 + r + hl * 8;
        if (m < M) {
          float v = acc[sm][sn][r] + bv;
          if (gelu) v = 0.5f * v * (1.f + erff(v * 0.70710678118f));
          size_t o = (size_t)m * N + n;
          if (residual) v += residual[o];
          if (outF32) outF32[o] = v;
          if (outF16) outF16[o] = (f16)v;
        }
      }
    }
  }
}

// ------------------------------------------- split QKV into padded head bufs
// qkv[M, 3*1024] -> q[b,h,np,d]*0.125, k[b,h,np,d], vT[b,h,d,np]; rows >=197 zero
__global__ __launch_bounds__(256) void split_qkv_kernel(
    const f16* __restrict__ qkv, f16* __restrict__ qp,
    f16* __restrict__ kp, f16* __restrict__ vT) {
  int idx = blockIdx.x * 256 + threadIdx.x;        // < 64*16*224*64
  int d   = idx & 63;
  int tok = (idx >> 6) % NPAD;
  int h   = (idx / (64 * NPAD)) & 15;
  int b   = idx / (64 * NPAD * 16);
  f16 qv = (f16)0.f, kv = (f16)0.f, vv = (f16)0.f;
  if (tok < NTOK) {
    const f16* base = qkv + (size_t)(b * NTOK + tok) * (3 * DMODEL) + h * 64 + d;
    qv = (f16)((float)base[0] * 0.125f);           // HEAD_DIM^-0.5
    kv = base[DMODEL];
    vv = base[2 * DMODEL];
  }
  size_t bh = (size_t)(b * NHEAD + h);
  qp[(bh * NPAD + tok) * HDIM + d] = qv;
  kp[(bh * NPAD + tok) * HDIM + d] = kv;
  vT[(bh * HDIM + d) * NPAD + tok] = vv;
}

// -------------------------------------------------- attention (one wave/blk)
__device__ __forceinline__ int rel_index(int r, int c) {
  if (r == 0 && c == 0) return 731;   // NUM_REL-1
  if (r == 0) return 729;             // NUM_REL-3
  if (c == 0) return 730;             // NUM_REL-2
  int pi = r - 1, qi = c - 1;
  int di = (pi / 14) - (qi / 14) + 13;
  int dj = (pi % 14) - (qi % 14) + 13;
  return di * 27 + dj;
}

__global__ __launch_bounds__(32) void attn_kernel(
    const f16* __restrict__ qp, const f16* __restrict__ kp,
    const f16* __restrict__ vT, const float* __restrict__ relTable,
    f16* __restrict__ o) {
  __shared__ f16 P[16 * NPAD];
  const int lane = threadIdx.x;
  const int hl = lane >> 4, li = lane & 15;
  int blk = blockIdx.x;
  const int qt = blk % 13; blk /= 13;              // query tile (13 x 16 = 208)
  const int h  = blk & 15;
  const int b  = blk >> 4;
  const size_t bh = (size_t)(b * NHEAD + h);
  const f16* Q  = qp + bh * NPAD * HDIM;
  const f16* Km = kp + bh * NPAD * HDIM;
  const f16* Vm = vT + bh * HDIM * NPAD;

  // Q fragments for this 16x64 slab (2 K-chunks of 32)
  v16h qf[2];
  {
    const f16* p = Q + (qt * 16 + li) * HDIM + hl * 8;
    qf[0] = mk16(*(const v8h*)p,        *(const v8h*)(p + 16));
    qf[1] = mk16(*(const v8h*)(p + 32), *(const v8h*)(p + 48));
  }
  // S = Q K^T : 13 key tiles
  v8f S[13];
#pragma unroll
  for (int j = 0; j < 13; ++j) {
    const f16* p = Km + (j * 16 + li) * HDIM + hl * 8;
    v16h k0 = mk16(*(const v8h*)p,        *(const v8h*)(p + 16));
    v16h k1 = mk16(*(const v8h*)(p + 32), *(const v8h*)(p + 48));
    v8f acc = {};
    acc = wmma_f16(qf[0], k0, acc);
    acc = wmma_f16(qf[1], k1, acc);
    S[j] = acc;
  }
  // relative-position bias + padding mask
#pragma unroll
  for (int j = 0; j < 13; ++j) {
    int col = j * 16 + li;
#pragma unroll
    for (int r = 0; r < 8; ++r) {
      int row = qt * 16 + r + hl * 8;
      float s = S[j][r];
      if (col < NTOK && row < NTOK)
        s += relTable[rel_index(row, col) * NHEAD + h];
      else
        s = -1e30f;
      S[j][r] = s;
    }
  }
  // softmax: each row lives across the 16 lanes of a half-wave
#pragma unroll
  for (int r = 0; r < 8; ++r) {
    float mx = -1e30f;
#pragma unroll
    for (int j = 0; j < 13; ++j) mx = fmaxf(mx, S[j][r]);
#pragma unroll
    for (int m = 1; m < 16; m <<= 1) mx = fmaxf(mx, __shfl_xor(mx, m, 16));
    float sum = 0.f;
#pragma unroll
    for (int j = 0; j < 13; ++j) {
      float e = __expf(S[j][r] - mx);
      S[j][r] = e; sum += e;
    }
#pragma unroll
    for (int m = 1; m < 16; m <<= 1) sum += __shfl_xor(sum, m, 16);
    float inv = 1.f / sum;
#pragma unroll
    for (int j = 0; j < 13; ++j) S[j][r] *= inv;
  }
  // spill probabilities to LDS row-major [16][NPAD], pad tile zeroed
#pragma unroll
  for (int j = 0; j < 13; ++j)
#pragma unroll
    for (int r = 0; r < 8; ++r)
      P[(r + hl * 8) * NPAD + j * 16 + li] = (f16)S[j][r];
  for (int i = lane; i < 16 * 16; i += 32)
    P[(i >> 4) * NPAD + 208 + (i & 15)] = (f16)0.f;
  __syncthreads();
  // O = P x V : 7 K-chunks of 32 over 224, 4 output d-tiles
  v8f oa[4] = {};
#pragma unroll
  for (int cc = 0; cc < 7; ++cc) {
    const f16* pp = P + li * NPAD + cc * 32 + hl * 8;
    v16h pf = mk16(*(const v8h*)pp, *(const v8h*)(pp + 16));
#pragma unroll
    for (int nt = 0; nt < 4; ++nt) {
      const f16* vp = Vm + (nt * 16 + li) * NPAD + cc * 32 + hl * 8;
      v16h vf = mk16(*(const v8h*)vp, *(const v8h*)(vp + 16));
      oa[nt] = wmma_f16(pf, vf, oa[nt]);
    }
  }
#pragma unroll
  for (int nt = 0; nt < 4; ++nt)
#pragma unroll
    for (int r = 0; r < 8; ++r) {
      int tok = qt * 16 + r + hl * 8;
      if (tok < NTOK)
        o[(size_t)(b * NTOK + tok) * DMODEL + h * HDIM + nt * 16 + li] =
            (f16)oa[nt][r];
    }
}

// ------------------------------------------------------------------ launcher
extern "C" void kernel_launch(void* const* d_in, const int* in_sizes, int n_in,
                              void* d_out, int out_size, void* d_ws,
                              size_t ws_size, hipStream_t stream) {
  const float* x      = (const float*)d_in[0];
  const float* qkv_w  = (const float*)d_in[1];
  const float* q_bias = (const float*)d_in[2];
  const float* v_bias = (const float*)d_in[3];
  const float* relTab = (const float*)d_in[4];
  const float* proj_w = (const float*)d_in[5];
  const float* proj_b = (const float*)d_in[6];
  const float* ln1_g  = (const float*)d_in[7];
  const float* ln1_b  = (const float*)d_in[8];
  const float* ln2_g  = (const float*)d_in[9];
  const float* ln2_b  = (const float*)d_in[10];
  const float* fc1_w  = (const float*)d_in[11];
  const float* fc1_b  = (const float*)d_in[12];
  const float* fc2_w  = (const float*)d_in[13];
  const float* fc2_b  = (const float*)d_in[14];

  char* ws = (char*)d_ws;
  size_t off = 0;
  auto alloc = [&](size_t bytes) -> char* {
    char* p = ws + off;
    off = (off + bytes + 255) & ~(size_t)255;
    return p;
  };
  f16*   wqkv16  = (f16*)alloc((size_t)3 * DMODEL * DMODEL * 2);
  f16*   wproj16 = (f16*)alloc((size_t)DMODEL * DMODEL * 2);
  f16*   wfc116  = (f16*)alloc((size_t)DFF * DMODEL * 2);
  f16*   wfc216  = (f16*)alloc((size_t)DMODEL * DFF * 2);
  float* qkvb    = (float*)alloc((size_t)3 * DMODEL * 4);
  f16*   h16     = (f16*)alloc((size_t)MROWS * DMODEL * 2);
  float* x1      = (float*)alloc((size_t)MROWS * DMODEL * 4);
  f16*   o16     = (f16*)alloc((size_t)MROWS * DMODEL * 2);
  f16*   qkv16   = (f16*)alloc((size_t)MROWS * 3 * DMODEL * 2);
  f16*   qp      = (f16*)alloc((size_t)BATCH * NHEAD * NPAD * HDIM * 2);
  f16*   kp      = (f16*)alloc((size_t)BATCH * NHEAD * NPAD * HDIM * 2);
  f16*   vT      = (f16*)alloc((size_t)BATCH * NHEAD * NPAD * HDIM * 2);
  // FC1 activation (12608x4096 f16) reuses the dead qkv16+qp region.
  f16*   act16   = qkv16;

  // 1) fp16 weight copies + fused qkv bias vector
  f32_to_f16_kernel<<<1024, 256, 0, stream>>>(qkv_w, wqkv16, 3 * DMODEL * DMODEL);
  f32_to_f16_kernel<<<1024, 256, 0, stream>>>(proj_w, wproj16, DMODEL * DMODEL);
  f32_to_f16_kernel<<<1024, 256, 0, stream>>>(fc1_w, wfc116, DFF * DMODEL);
  f32_to_f16_kernel<<<1024, 256, 0, stream>>>(fc2_w, wfc216, DMODEL * DFF);
  qkv_bias_kernel<<<12, 256, 0, stream>>>(q_bias, v_bias, qkvb);

  // 2) LN1 -> f16
  ln_f16_kernel<<<MROWS, 256, 0, stream>>>(x, ln1_g, ln1_b, h16);
  // 3) QKV GEMM (+bias) -> f16
  gemm_wmma_kernel<<<dim3(48, 99), 256, 0, stream>>>(
      h16, wqkv16, qkvb, nullptr, nullptr, qkv16, MROWS, 3 * DMODEL, DMODEL, 0);
  // 4) split into padded per-head Q (scaled), K, V^T
  split_qkv_kernel<<<(BATCH * NHEAD * NPAD * HDIM) / 256, 256, 0, stream>>>(
      qkv16, qp, kp, vT);
  // 5) attention (S=QK^T + rel bias, softmax, O=PV)
  attn_kernel<<<BATCH * NHEAD * 13, 32, 0, stream>>>(qp, kp, vT, relTab, o16);
  // 6) proj GEMM + bias + residual(x) -> x1 fp32
  gemm_wmma_kernel<<<dim3(16, 99), 256, 0, stream>>>(
      o16, wproj16, proj_b, x, x1, nullptr, MROWS, DMODEL, DMODEL, 0);
  // 7) LN2 -> f16
  ln_f16_kernel<<<MROWS, 256, 0, stream>>>(x1, ln2_g, ln2_b, h16);
  // 8) FC1 GEMM + bias + exact GELU -> f16
  gemm_wmma_kernel<<<dim3(64, 99), 256, 0, stream>>>(
      h16, wfc116, fc1_b, nullptr, nullptr, act16, MROWS, DFF, DMODEL, 1);
  // 9) FC2 GEMM + bias + residual(x1) -> d_out fp32
  gemm_wmma_kernel<<<dim3(16, 99), 256, 0, stream>>>(
      act16, wfc216, fc2_b, x1, (float*)d_out, nullptr, MROWS, DMODEL, DFF, 0);
}
